// AutoEncoder_49160195670160
// MI455X (gfx1250) — compile-verified, matrix operand order checked by reading, not run on hardware
//
#include <hip/hip_runtime.h>
#include <hip/hip_bf16.h>
#include <stdint.h>

// ---------------------------------------------------------------------------
// Types for CDNA5 WMMA (wave32): bf16 16x16x32, f32 accumulate
// ---------------------------------------------------------------------------
typedef __attribute__((ext_vector_type(16))) __bf16 v16bf;
typedef __attribute__((ext_vector_type(8)))  __bf16 v8bf;
typedef __attribute__((ext_vector_type(8)))  float  v8f;

#define TERN_EPS 1e-8f

#ifndef __has_builtin
#define __has_builtin(x) 0
#endif

// ---------------------------------------------------------------------------
// CDNA5 async global -> LDS copy (ASYNCcnt path), with inline-asm fallback
// Toolchain-confirmed signature: (int4 AS1* gaddr, int4 AS3* lds, imm, imm)
// ---------------------------------------------------------------------------
__device__ __forceinline__ void async_copy_b128(void* lds, const void* g) {
#if __has_builtin(__builtin_amdgcn_global_load_async_to_lds_b128)
  typedef __attribute__((ext_vector_type(4))) int v4i;
  typedef __attribute__((address_space(1))) v4i* gv4p;
  typedef __attribute__((address_space(3))) v4i* lv4p;
  __builtin_amdgcn_global_load_async_to_lds_b128(
      (gv4p)(__attribute__((address_space(1))) void*)g,
      (lv4p)(__attribute__((address_space(3))) void*)lds, 0, 0);
#else
  asm volatile("global_load_async_to_lds_b128 %0, %1, off"
               :
               : "v"((unsigned int)(uintptr_t)lds),
                 "v"((unsigned long long)(uintptr_t)g)
               : "memory");
#endif
}

__device__ __forceinline__ void wait_asynccnt0() {
#if __has_builtin(__builtin_amdgcn_s_wait_asynccnt)
  __builtin_amdgcn_s_wait_asynccnt(0);
#else
  asm volatile("s_wait_asynccnt 0x0" ::: "memory");
#endif
}

__device__ __forceinline__ unsigned short f2bf16(float f) {
  // round-to-nearest-even float -> bf16 bits
  unsigned int u = __builtin_bit_cast(unsigned int, f);
  unsigned int r = (u + 0x7FFFu + ((u >> 16) & 1u)) >> 16;
  return (unsigned short)r;
}

// ---------------------------------------------------------------------------
// Scale reduction: pass 1, per-weight partial sums of |w|
// ---------------------------------------------------------------------------
__global__ void abs_sum_part(const float* __restrict__ w, long long n,
                             float* __restrict__ part) {
  __shared__ float sm[256];
  float a = 0.0f;
  for (long long i = (long long)blockIdx.x * 256 + threadIdx.x; i < n;
       i += (long long)gridDim.x * 256)
    a += fabsf(w[i]);
  sm[threadIdx.x] = a;
  __syncthreads();
  for (int s = 128; s > 0; s >>= 1) {
    if ((int)threadIdx.x < s) sm[threadIdx.x] += sm[threadIdx.x + s];
    __syncthreads();
  }
  if (threadIdx.x == 0) part[blockIdx.x] = sm[0];
}

struct Numels { long long n[8]; };

// pass 2: one block per weight -> scales[i] = sum/numel
__global__ void finalize_scales(const float* __restrict__ part,
                                float* __restrict__ scales, Numels ne) {
  __shared__ float sm[256];
  const int i = blockIdx.x, t = threadIdx.x;
  float a = part[i * 512 + t] + part[i * 512 + 256 + t];
  sm[t] = a;
  __syncthreads();
  for (int s = 128; s > 0; s >>= 1) {
    if (t < s) sm[t] += sm[t + s];
    __syncthreads();
  }
  if (t == 0) scales[i] = sm[0] / (float)ne.n[i];
}

// ---------------------------------------------------------------------------
// x (fp32) -> bf16, 4-wide
// ---------------------------------------------------------------------------
__global__ void cvt_f32_to_bf16(const float* __restrict__ x,
                                unsigned short* __restrict__ o, int n) {
  int i = (blockIdx.x * 256 + threadIdx.x) * 4;
  if (i < n) {
    float4 v = *(const float4*)(x + i);
    ushort4 r;
    r.x = f2bf16(v.x); r.y = f2bf16(v.y); r.z = f2bf16(v.z); r.w = f2bf16(v.w);
    *(ushort4*)(o + i) = r;
  }
}

// ---------------------------------------------------------------------------
// One-shot per-layer weight quantization: fp32 -> ternary {-1,0,+1} bf16 bits.
// (s is applied in fp32 in the GEMM epilogue, so this is lossless ternary.)
// ---------------------------------------------------------------------------
__global__ void quant_w(const float* __restrict__ w,
                        const float* __restrict__ scale_p,
                        unsigned short* __restrict__ o, int n) {
  const float inv = 1.0f / (scale_p[0] + TERN_EPS);
  int i = (blockIdx.x * 256 + threadIdx.x) * 4;
  if (i < n) {
    float4 v = *(const float4*)(w + i);
    ushort4 t;
    float r;
    r = rintf(v.x * inv); t.x = r > 0.5f ? 0x3F80u : (r < -0.5f ? 0xBF80u : 0u);
    r = rintf(v.y * inv); t.y = r > 0.5f ? 0x3F80u : (r < -0.5f ? 0xBF80u : 0u);
    r = rintf(v.z * inv); t.z = r > 0.5f ? 0x3F80u : (r < -0.5f ? 0xBF80u : 0u);
    r = rintf(v.w * inv); t.w = r > 0.5f ? 0x3F80u : (r < -0.5f ? 0xBF80u : 0u);
    *(ushort4*)(o + i) = t;
  }
}

// ---------------------------------------------------------------------------
// Ternary GEMM:  out[M,N] = (Abf16[M,K] @ Wq[N,K]^T) * s + bias
// Block tile 128x128x32, 256 threads = 8 waves (wave32), wave grid 4x2,
// each wave: 2x4 fragments of v_wmma_f32_16x16x32_bf16.
// Double-buffered LDS; BOTH tiles staged with async global->LDS (ASYNCcnt):
// steady-state loop has zero data-movement VALU.
// ---------------------------------------------------------------------------
#define BM 128
#define BN 128
#define BK 32
#define LDSS 40  // padded LDS row stride in bf16 elements (80B -> conflict-free)

__global__ void __launch_bounds__(256)
ternary_gemm_bf16(const unsigned short* __restrict__ A,   // bf16 [M,K]
                  const unsigned short* __restrict__ Wq,  // ternary bf16 [N,K]
                  const float* __restrict__ bias,         // fp32 [N]
                  const float* __restrict__ scale_p,      // &s
                  int K, int N,
                  unsigned short* __restrict__ outB,      // bf16 [M,N]
                  float* __restrict__ outF,               // fp32 [M,N] or null
                  int relu) {
  __shared__ unsigned short As[2][BM * LDSS];
  __shared__ unsigned short Bs[2][BN * LDSS];

  const int tid  = threadIdx.x;
  const int lane = tid & 31;
  const int wid  = tid >> 5;
  const int wave_m = wid >> 1;  // 0..3 -> 32 rows each
  const int wave_n = wid & 1;   // 0..1 -> 64 cols each

  const int bm = blockIdx.y * BM;
  const int bn = blockIdx.x * BN;

  const float s = scale_p[0];

  // tile staging: 512 chunks of 8 bf16 (16B) per tile; 2 chunks/thread/tile
  const int ar0 = tid >> 2,           akb0 = (tid & 3) * 8;
  const int ar1 = (tid + 256) >> 2,   akb1 = ((tid + 256) & 3) * 8;

  // Fragment LDS offsets (ISA 7.12.2 bf16 striping):
  //   lanes 0-15: row=lane,     K 0..7 then 16..23
  //   lanes16-31: row=lane-16,  K 8..15 then 24..31
  const int lrow  = lane & 15;
  const int khalf = (lane >> 4) * 8;

  int aoff[2], boff[4];
#pragma unroll
  for (int tm = 0; tm < 2; ++tm)
    aoff[tm] = (wave_m * 32 + tm * 16 + lrow) * LDSS + khalf;
#pragma unroll
  for (int tn = 0; tn < 4; ++tn)
    boff[tn] = (wave_n * 64 + tn * 16 + lrow) * LDSS + khalf;

  v8f acc[2][4] = {};

  const unsigned short* gA = A  + (size_t)bm * K;
  const unsigned short* gW = Wq + (size_t)bn * K;

  // stage tile at k0 into LDS stage st (all async, no VGPR round trip)
  auto stage = [&](int k0, int st) {
    async_copy_b128(&As[st][ar0 * LDSS + akb0], gA + (size_t)ar0 * K + k0 + akb0);
    async_copy_b128(&As[st][ar1 * LDSS + akb1], gA + (size_t)ar1 * K + k0 + akb1);
    async_copy_b128(&Bs[st][ar0 * LDSS + akb0], gW + (size_t)ar0 * K + k0 + akb0);
    async_copy_b128(&Bs[st][ar1 * LDSS + akb1], gW + (size_t)ar1 * K + k0 + akb1);
  };

  // prologue: stage first tile
  stage(0, 0);
  wait_asynccnt0();
  __syncthreads();

  int cur = 0;
  for (int k0 = 0; k0 < K; k0 += BK) {
    const int nxt = cur ^ 1;
    // stage next tile while computing current one
    if (k0 + BK < K) stage(k0 + BK, nxt);

    const unsigned short* as = As[cur];
    const unsigned short* bs = Bs[cur];

    v16bf afrag[2], bfrag[4];
#pragma unroll
    for (int tm = 0; tm < 2; ++tm) {
      v8bf lo = *(const v8bf*)(as + aoff[tm]);
      v8bf hi = *(const v8bf*)(as + aoff[tm] + 16);
      afrag[tm] = __builtin_shufflevector(lo, hi, 0, 1, 2, 3, 4, 5, 6, 7, 8, 9,
                                          10, 11, 12, 13, 14, 15);
    }
#pragma unroll
    for (int tn = 0; tn < 4; ++tn) {
      v8bf lo = *(const v8bf*)(bs + boff[tn]);
      v8bf hi = *(const v8bf*)(bs + boff[tn] + 16);
      bfrag[tn] = __builtin_shufflevector(lo, hi, 0, 1, 2, 3, 4, 5, 6, 7, 8, 9,
                                          10, 11, 12, 13, 14, 15);
    }
#pragma unroll
    for (int tm = 0; tm < 2; ++tm)
#pragma unroll
      for (int tn = 0; tn < 4; ++tn)
        acc[tm][tn] = __builtin_amdgcn_wmma_f32_16x16x32_bf16(
            false, afrag[tm], false, bfrag[tn], (short)0, acc[tm][tn], false,
            false);

    // next stage's async data must be fully landed for all waves, and all
    // waves must be done reading `cur` before it is overwritten
    wait_asynccnt0();
    __syncthreads();
    cur = nxt;
  }

  // ---- epilogue: *s + bias, optional relu, fp32 + bf16 stores ----
  // C/D layout: lanes0-15 -> M=vgpr, N=lane; lanes16-31 -> M=vgpr+8, N=lane-16
#pragma unroll
  for (int tn = 0; tn < 4; ++tn) {
    const int col   = bn + wave_n * 64 + tn * 16 + lrow;
    const float bcl = bias[col];
#pragma unroll
    for (int tm = 0; tm < 2; ++tm) {
      const int row0 = bm + wave_m * 32 + tm * 16 + khalf;
#pragma unroll
      for (int i = 0; i < 8; ++i) {
        float v = acc[tm][tn][i] * s + bcl;
        if (relu) v = fmaxf(v, 0.0f);
        const size_t idx = (size_t)(row0 + i) * N + col;
        if (outF) outF[idx] = v;
        outB[idx] = f2bf16(v);
      }
    }
  }
}

// ---------------------------------------------------------------------------
// mean(exp(xc)) : two-pass deterministic reduction
// ---------------------------------------------------------------------------
__global__ void exp_part(const float* __restrict__ x, long long n,
                         float* __restrict__ part) {
  __shared__ float sm[256];
  float a = 0.0f;
  for (long long i = (long long)blockIdx.x * 256 + threadIdx.x; i < n;
       i += (long long)gridDim.x * 256)
    a += expf(x[i]);
  sm[threadIdx.x] = a;
  __syncthreads();
  for (int s = 128; s > 0; s >>= 1) {
    if ((int)threadIdx.x < s) sm[threadIdx.x] += sm[threadIdx.x + s];
    __syncthreads();
  }
  if (threadIdx.x == 0) part[blockIdx.x] = sm[0];
}

__global__ void exp_final(const float* __restrict__ part,
                          float* __restrict__ out, float count) {
  __shared__ float sm[256];
  const int t = threadIdx.x;
  float a = part[t] + part[t + 256] + part[t + 512] + part[t + 768];
  sm[t] = a;
  __syncthreads();
  for (int s = 128; s > 0; s >>= 1) {
    if (t < s) sm[t] += sm[t + s];
    __syncthreads();
  }
  if (t == 0) out[0] = sm[0] / count;
}

// ---------------------------------------------------------------------------
// Host-side orchestration
// d_in: 0:x 1:ew1 2:ew2 3:ew3 4:ew4 5:dw1 6:dw2 7:dw3 8:dw4
//       9:eb1 10:eb2 11:eb3 12:eb4 13:db1 14:db2 15:db3 16:db4
// d_out: xc[4096*1024] | z[4096*2048] | loss[1]
// ---------------------------------------------------------------------------
extern "C" void kernel_launch(void* const* d_in, const int* in_sizes, int n_in,
                              void* d_out, int out_size, void* d_ws,
                              size_t ws_size, hipStream_t stream) {
  (void)in_sizes; (void)n_in; (void)out_size; (void)ws_size;

  const float* x = (const float*)d_in[0];
  const float* Wt[8];
  const float* Bi[8];
  for (int i = 0; i < 8; ++i) {
    Wt[i] = (const float*)d_in[1 + i];
    Bi[i] = (const float*)d_in[9 + i];
  }

  char* ws = (char*)d_ws;
  float* scales = (float*)ws;                       //   8 floats
  float* spart  = (float*)(ws + 256);               //   8*512 floats
  float* epart  = (float*)(ws + 256 + 8 * 512 * 4); //   1024 floats
  unsigned short* bufA = (unsigned short*)(ws + 32768);   // 4096*4096 bf16
  unsigned short* bufB = bufA + (size_t)4096 * 4096;      // 4096*4096 bf16
  unsigned short* wq   = bufB + (size_t)4096 * 4096;      // 4096*4096 bf16

  static const long long NM[8] = {8388608LL,  16777216LL, 16777216LL,
                                  4194304LL,  4194304LL,  16777216LL,
                                  16777216LL, 8388608LL};
  Numels ne;
  for (int i = 0; i < 8; ++i) ne.n[i] = NM[i];

  // per-tensor scales s = mean|w|
  for (int i = 0; i < 8; ++i)
    abs_sum_part<<<512, 256, 0, stream>>>(Wt[i], NM[i], spart + i * 512);
  finalize_scales<<<8, 256, 0, stream>>>(spart, scales, ne);

  // x -> bf16
  cvt_f32_to_bf16<<<8192, 256, 0, stream>>>(x, bufA, 4096 * 2048);

  float* xcF   = (float*)d_out;
  float* zF    = xcF + (size_t)4096 * 1024;
  float* lossF = zF + (size_t)4096 * 2048;

  // per-layer: quantize weights once (ternary bf16), then async-WMMA GEMM
  struct L { const unsigned short* a; int K, N; unsigned short* ob; float* of; int relu; };
  const L lay[8] = {
      {bufA, 2048, 4096, bufB, nullptr, 0},  // ew1
      {bufB, 4096, 4096, bufA, nullptr, 0},  // ew2
      {bufA, 4096, 4096, bufB, nullptr, 0},  // ew3
      {bufB, 4096, 1024, bufA, xcF,     1},  // ew4 + relu -> xc
      {bufA, 1024, 4096, bufB, nullptr, 0},  // dw1
      {bufB, 4096, 4096, bufA, nullptr, 0},  // dw2
      {bufA, 4096, 4096, bufB, nullptr, 0},  // dw3
      {bufB, 4096, 2048, bufA, zF,      0},  // dw4 -> z
  };

  for (int l = 0; l < 8; ++l) {
    quant_w<<<(unsigned)(NM[l] / 1024), 256, 0, stream>>>(Wt[l], scales + l, wq,
                                                          (int)NM[l]);
    ternary_gemm_bf16<<<dim3(lay[l].N / 128, 32), 256, 0, stream>>>(
        lay[l].a, wq, Bi[l], scales + l, lay[l].K, lay[l].N, lay[l].ob,
        lay[l].of, lay[l].relu);
    if (l == 3) {  // sparse loss = mean(exp(xc))
      exp_part<<<1024, 256, 0, stream>>>(xcF, 4194304LL, epart);
      exp_final<<<1, 256, 0, stream>>>(epart, lossF, 4194304.0f);
    }
  }
}
